// MoE_86835648791008
// MI455X (gfx1250) — matrix operand order repaired and checked by reading, not compiled, not run on hardware
//
#include <hip/hip_runtime.h>
#include <cmath>

// ---------------- types for WMMA ----------------
typedef __bf16 bf16_t;
typedef __attribute__((ext_vector_type(16))) __bf16 v16bf;
typedef __attribute__((ext_vector_type(8)))  float  v8f;

union V16 { v16bf v; uint4 u[2]; };

__device__ __forceinline__ v8f wmma_bf16(v16bf a, v16bf b, v8f c) {
  // D = A(16x32) * B(32x16) + C, f32 accumulate
  return __builtin_amdgcn_wmma_f32_16x16x32_bf16(false, a, false, b, (short)0, c, false, false);
}

// A-fragment: lane (m = lane&15). elems j0..7 -> K = kb*32 + half8 + j,
//             elems j8..15 -> K = kb*32 + 16 + half8 + (j-8). half8 = 8*(lane>=16).
__device__ __forceinline__ v16bf frag_a(const bf16_t* row, int kb, int half8) {
  V16 r;
  const bf16_t* p = row + kb * 32 + half8;
  r.u[0] = *reinterpret_cast<const uint4*>(p);
  r.u[1] = *reinterpret_cast<const uint4*>(p + 16);
  return r.v;
}
// B-fragment: lane (n = lane&15). elems j0..7 -> K = kb*32 + half16/2 + j,
//             elems j8..15 -> K = kb*32 + 16 + half16/2 + (j-8). half16 = 16*(lane>=16).
__device__ __forceinline__ v16bf frag_b(const bf16_t* row, int kb, int half16) {
  V16 r;
  const bf16_t* p = row + kb * 32 + (half16 >> 1);
  r.u[0] = *reinterpret_cast<const uint4*>(p);
  r.u[1] = *reinterpret_cast<const uint4*>(p + 16);
  return r.v;
}

// ---------------- problem constants ----------------
#define NUM_EXPERTS 8
#define TOKENS      2048     // B*S
#define DIM         1024
#define FFN         4096
#define OUT_ELEMS   (TOKENS * DIM)   // 2097152, then +1 for cv

// workspace layout (bytes)
static constexpr size_t GS_OFF   = 0;                    // gate_scores: 2048*8 f32
static constexpr size_t CNT_OFF  = 65536;                // 8 i32
static constexpr size_t OFF_OFF  = CNT_OFF + 32;         // 8 i32 (padded row offsets)
static constexpr size_t TOK_OFF  = OFF_OFF + 32;         // 8*2048 i32
static constexpr size_t PROB_OFF = TOK_OFF + 65536;      // 8*2048 f32
static constexpr size_t IMP_OFF  = PROB_OFF + 65536;     // 16 f32
static constexpr size_t H_OFF    = 262144;               // H: 4224 x 4096 bf16 (~33 MB)

// LDS row strides (elements), padded to kill bank conflicts; byte stride stays 16B-aligned
#define S1 1032   // pass1: 1024 + 8
#define S2 4104   // pass2: 4096 + 8

// ================= router =================
__global__ void __launch_bounds__(256) router_kernel(
    const float* __restrict__ x, const float* __restrict__ eps,
    const float* __restrict__ gw, const float* __restrict__ gb,
    const float* __restrict__ nw, const float* __restrict__ nb,
    float* __restrict__ gs, int* __restrict__ cnt,
    int* __restrict__ tokL, float* __restrict__ probL) {
  const int lane = threadIdx.x & 31;
  const int wid  = threadIdx.x >> 5;
  const int t    = blockIdx.x * 8 + wid;     // exact: 256 blocks * 8 waves = 2048

  float xr[32];
  const float* xp = x + (size_t)t * DIM;
#pragma unroll
  for (int i = 0; i < 32; ++i) xr[i] = xp[lane + i * 32];

  float g[NUM_EXPERTS], npre[NUM_EXPERTS];
#pragma unroll
  for (int e = 0; e < NUM_EXPERTS; ++e) {
    const float* gwp = gw + e * DIM;
    const float* nwp = nw + e * DIM;
    float ag = 0.f, an = 0.f;
#pragma unroll
    for (int i = 0; i < 32; ++i) {
      float xv = xr[i];
      ag = fmaf(xv, gwp[lane + i * 32], ag);
      an = fmaf(xv, nwp[lane + i * 32], an);
    }
#pragma unroll
    for (int m = 16; m >= 1; m >>= 1) {
      ag += __shfl_xor(ag, m, 32);
      an += __shfl_xor(an, m, 32);
    }
    g[e]    = ag + gb[e];
    npre[e] = an + nb[e];
  }

  float nl[NUM_EXPERTS];
#pragma unroll
  for (int e = 0; e < NUM_EXPERTS; ++e) {
    float s  = npre[e];
    float sp = (s > 20.f) ? s : log1pf(expf(s));    // stable softplus
    nl[e] = g[e] + eps[(size_t)t * NUM_EXPERTS + e] * sp;
  }

  // top-2 (first occurrence wins ties, matching lax.top_k)
  int i1 = 0; float v1 = nl[0];
#pragma unroll
  for (int e = 1; e < NUM_EXPERTS; ++e) if (nl[e] > v1) { v1 = nl[e]; i1 = e; }
  int i2 = (i1 == 0) ? 1 : 0; float v2 = nl[i2];
#pragma unroll
  for (int e = 0; e < NUM_EXPERTS; ++e)
    if (e != i1 && e != i2 && nl[e] > v2) { v2 = nl[e]; i2 = e; }

  float d  = expf(v2 - v1);                 // <= 1, stable softmax over the pair
  float p1 = 1.f / (1.f + d);
  float p2 = d * p1;

  if (lane < NUM_EXPERTS) {
    float p = (lane == i1) ? p1 : ((lane == i2) ? p2 : 0.f);
    gs[(size_t)t * NUM_EXPERTS + lane] = p;
  }
  if (lane == 0) {
    int s1 = atomicAdd(&cnt[i1], 1);
    tokL [i1 * TOKENS + s1] = t;
    probL[i1 * TOKENS + s1] = p1;
    int s2 = atomicAdd(&cnt[i2], 1);
    tokL [i2 * TOKENS + s2] = t;
    probL[i2 * TOKENS + s2] = p2;
  }
}

// ================= finalize: importance sums, aux CV loss, padded offsets =================
__global__ void finalize_kernel(const float* __restrict__ gs, const int* __restrict__ cnt,
                                int* __restrict__ off, float* __restrict__ imp,
                                float* __restrict__ cv_out) {
  const int tid = threadIdx.x;
  if (tid < 16) {                           // deterministic fixed-order sums
    int b = tid >> 3, e = tid & 7;
    float s = 0.f;
    for (int i = 0; i < 1024; ++i) s += gs[(size_t)(b * 1024 + i) * NUM_EXPERTS + e];
    imp[tid] = s;
  }
  __syncthreads();
  if (tid == 0) {
    float mean = 0.f;
    for (int i = 0; i < 16; ++i) mean += imp[i];
    mean *= (1.f / 16.f);
    float var = 0.f;
    for (int i = 0; i < 16; ++i) { float dd = imp[i] - mean; var += dd * dd; }
    var *= (1.f / 15.f);                    // ddof=1
    *cv_out = var / (mean * mean) * 0.01f;
    int o = 0;
    for (int e = 0; e < NUM_EXPERTS; ++e) { off[e] = o; o += ((cnt[e] + 15) >> 4) << 4; }
  }
}

// ================= pass 1: H = silu(X W1^T) * (X W3^T)  (per expert, gathered tokens) ==========
// grid (64 ffn-chunks, 8 experts), 128 threads (4 waves), dyn LDS = 144*S1*2 bytes (~290 KB)
__global__ void __launch_bounds__(128) ffn13_kernel(
    const float* __restrict__ x, const float* __restrict__ w1g, const float* __restrict__ w3g,
    const int* __restrict__ cnt, const int* __restrict__ off,
    const int* __restrict__ tokL, bf16_t* __restrict__ H) {
  extern __shared__ bf16_t smem[];          // rows 0..63: w1 chunk, 64..127: w3 chunk, 128..143: X tile
  const int e      = blockIdx.y;
  const int nBase  = blockIdx.x * 64;
  const int count  = cnt[e];
  const int nTiles = (count + 15) >> 4;
  if (nTiles == 0) return;
  const int tid = threadIdx.x;

  // stream weight chunks HBM->LDS once (fp32 -> bf16)
  const float* w1p = w1g + ((size_t)e * FFN + nBase) * DIM;
  const float* w3p = w3g + ((size_t)e * FFN + nBase) * DIM;
  for (int i = tid; i < 64 * 256; i += 128) {
    int idx = i * 4, row = idx >> 10, col = idx & 1023;
    float4 a = *reinterpret_cast<const float4*>(w1p + (size_t)row * DIM + col);
    bf16_t* d1 = smem + row * S1 + col;
    d1[0] = (bf16_t)a.x; d1[1] = (bf16_t)a.y; d1[2] = (bf16_t)a.z; d1[3] = (bf16_t)a.w;
    float4 b = *reinterpret_cast<const float4*>(w3p + (size_t)row * DIM + col);
    bf16_t* d3 = smem + (64 + row) * S1 + col;
    d3[0] = (bf16_t)b.x; d3[1] = (bf16_t)b.y; d3[2] = (bf16_t)b.z; d3[3] = (bf16_t)b.w;
  }

  const int lane   = tid & 31;
  const int wid    = tid >> 5;              // 4 waves -> 4 x 16 ffn rows
  const int c16    = lane & 15;
  const int half8  = (lane >> 4) << 3;
  const int half16 = (lane >> 4) << 4;
  const int hoffE  = off[e];
  const bf16_t* arow  = smem + (128 + c16) * S1;
  const bf16_t* b1row = smem + (wid * 16 + c16) * S1;
  const bf16_t* b3row = smem + (64 + wid * 16 + c16) * S1;

  for (int tile = 0; tile < nTiles; ++tile) {
    const int tileBase = tile * 16;
    __syncthreads();
    // gather X tile (fp32 -> bf16); padded rows -> 0 so H pad rows come out exactly 0
    for (int i = tid; i < 16 * 256; i += 128) {
      int idx = i * 4, r = idx >> 10, col = idx & 1023;
      int slot = tileBase + r;
      float4 a = make_float4(0.f, 0.f, 0.f, 0.f);
      if (slot < count)
        a = *reinterpret_cast<const float4*>(x + (size_t)tokL[e * TOKENS + slot] * DIM + col);
      bf16_t* d = smem + (128 + r) * S1 + col;
      d[0] = (bf16_t)a.x; d[1] = (bf16_t)a.y; d[2] = (bf16_t)a.z; d[3] = (bf16_t)a.w;
    }
    __syncthreads();

    v8f aw = {}; v8f av = {};
#pragma unroll 4
    for (int kb = 0; kb < 32; ++kb) {       // K = 1024 = 32 * 32
      v16bf A  = frag_a(arow,  kb, half8);
      v16bf B1 = frag_b(b1row, kb, half16);
      v16bf B3 = frag_b(b3row, kb, half16);
      aw = wmma_bf16(A, B1, aw);
      av = wmma_bf16(A, B3, av);
    }
    bf16_t* hp = H + (size_t)(hoffE + tileBase) * FFN + nBase + wid * 16 + c16;
#pragma unroll
    for (int v = 0; v < 8; ++v) {
      int m = v + half8;
      float s = aw[v];
      float h = (s / (1.f + expf(-s))) * av[v];   // silu(s) * v
      hp[(size_t)m * FFN] = (bf16_t)h;
    }
  }
}

// ================= pass 2: out += p * (H W2^T) =================
// grid (32 dim-chunks, 8 experts), 64 threads (2 waves), dyn LDS = 32*S2*2 bytes (~263 KB)
__global__ void __launch_bounds__(64) ffn2_kernel(
    const float* __restrict__ w2g, const bf16_t* __restrict__ H,
    const int* __restrict__ cnt, const int* __restrict__ off,
    const int* __restrict__ tokL, const float* __restrict__ probL,
    float* __restrict__ out) {
  extern __shared__ bf16_t smem[];          // rows 0..31: w2 chunk [32 x 4096]
  const int e      = blockIdx.y;
  const int dBase  = blockIdx.x * 32;
  const int count  = cnt[e];
  const int nTiles = (count + 15) >> 4;
  if (nTiles == 0) return;
  const int tid = threadIdx.x;

  const float* w2p = w2g + ((size_t)e * DIM + dBase) * FFN;
  for (int i = tid; i < 32 * 1024; i += 64) {
    int idx = i * 4, row = idx >> 12, col = idx & 4095;
    float4 a = *reinterpret_cast<const float4*>(w2p + (size_t)row * FFN + col);
    bf16_t* d = smem + row * S2 + col;
    d[0] = (bf16_t)a.x; d[1] = (bf16_t)a.y; d[2] = (bf16_t)a.z; d[3] = (bf16_t)a.w;
  }
  __syncthreads();

  const int lane   = tid & 31;
  const int wid    = tid >> 5;
  const int c16    = lane & 15;
  const int half8  = (lane >> 4) << 3;
  const int half16 = (lane >> 4) << 4;
  const int hoffE  = off[e];
  const bf16_t* brow = smem + (wid * 16 + c16) * S2;

  for (int tile = 0; tile < nTiles; ++tile) {
    const int tileBase = tile * 16;
    const bf16_t* arow = H + (size_t)(hoffE + tileBase + c16) * FFN;
    if (tile + 1 < nTiles)                  // global_prefetch_b8 for next A tile
      __builtin_prefetch(H + (size_t)(hoffE + tileBase + 16 + c16) * FFN, 0, 0);
    v8f acc = {};
#pragma unroll 4
    for (int kb = 0; kb < 128; ++kb) {      // K = FFN = 4096 = 128 * 32
      v16bf A = frag_a(arow, kb, half8);    // bf16 straight from H, no convert
      v16bf B = frag_b(brow, kb, half16);
      acc = wmma_bf16(A, B, acc);
    }
#pragma unroll
    for (int v = 0; v < 8; ++v) {
      int m = v + half8;
      int slot = tileBase + m;
      if (slot < count) {
        int   tok = tokL [e * TOKENS + slot];
        float p   = probL[e * TOKENS + slot];
        float* op = out + (size_t)tok * DIM + dBase + wid * 16 + c16;
        __hip_atomic_fetch_add(op, p * acc[v], __ATOMIC_RELAXED, __HIP_MEMORY_SCOPE_AGENT);
      }
    }
  }
}

// ================= launch =================
extern "C" void kernel_launch(void* const* d_in, const int* in_sizes, int n_in,
                              void* d_out, int out_size, void* d_ws, size_t ws_size,
                              hipStream_t stream) {
  (void)in_sizes; (void)n_in; (void)out_size; (void)ws_size;
  const float* x   = (const float*)d_in[0];
  const float* eps = (const float*)d_in[1];
  const float* gw  = (const float*)d_in[2];
  const float* gb  = (const float*)d_in[3];
  const float* nw  = (const float*)d_in[4];
  const float* nb  = (const float*)d_in[5];
  const float* w1  = (const float*)d_in[6];
  const float* w2  = (const float*)d_in[7];
  const float* w3  = (const float*)d_in[8];
  float* out = (float*)d_out;

  char* ws = (char*)d_ws;
  float*  gs    = (float*)(ws + GS_OFF);
  int*    cnt   = (int*)  (ws + CNT_OFF);
  int*    off   = (int*)  (ws + OFF_OFF);
  int*    tokL  = (int*)  (ws + TOK_OFF);
  float*  probL = (float*)(ws + PROB_OFF);
  float*  imp   = (float*)(ws + IMP_OFF);
  bf16_t* Hbuf  = (bf16_t*)(ws + H_OFF);

  hipMemsetAsync(ws + CNT_OFF, 0, 32, stream);
  hipMemsetAsync(d_out, 0, (size_t)OUT_ELEMS * sizeof(float), stream);

  router_kernel<<<TOKENS / 8, 256, 0, stream>>>(x, eps, gw, gb, nw, nb, gs, cnt, tokL, probL);
  finalize_kernel<<<1, 32, 0, stream>>>(gs, cnt, off, imp, out + OUT_ELEMS);
  ffn13_kernel<<<dim3(FFN / 64, NUM_EXPERTS), 128, (size_t)144 * S1 * sizeof(bf16_t), stream>>>(
      x, w1, w3, cnt, off, tokL, Hbuf);
  ffn2_kernel<<<dim3(DIM / 32, NUM_EXPERTS), 64, (size_t)32 * S2 * sizeof(bf16_t), stream>>>(
      w2, Hbuf, cnt, off, tokL, probL, out);
}
